// STMambaBlock_12094627905785
// MI455X (gfx1250) — matrix operand order repaired and checked by reading, not compiled, not run on hardware
//
#include <hip/hip_runtime.h>
#include <hip/hip_bf16.h>
#include <math.h>

// ---------------- problem constants ----------------
#define Bsz 8
#define Tt 8
#define Cc 128
#define Hh 32
#define Ww 32
#define Nn_TOK 1024           // H*W
#define Kk_TOK 512            // kept tokens (epoch 20 -> ratio 0.5)
#define DI 256                // d_inner
#define DS 16                 // d_state
#define DTR 8                 // dt_rank
#define NH 4
#define DH 32
#define ROWS (Bsz*Kk_TOK*Tt)  // 32768 mamba rows
#define BT (Bsz*Tt)           // 64 attention batches
#define XPAD 48               // x_proj N padded 40 -> 48

typedef __attribute__((ext_vector_type(16))) _Float16 v16h;
typedef __attribute__((ext_vector_type(8)))  float    v8f;
typedef __attribute__((ext_vector_type(4)))  unsigned int v4u;
typedef __attribute__((ext_vector_type(8)))  int      v8i;
typedef __attribute__((ext_vector_type(4)))  int      v4i;

union V16 { v16h h; uint4 q[2]; };

// A fragment: 16x32 f16, row-major source with leading dim lda (halves).
// lane l: row = mbase + (l&15); hi=l>>4; elements j<8 -> k=kbase+hi*8+j,
// j>=8 -> k=kbase+hi*8+16+(j-8). Two contiguous 16B loads.
static __device__ inline v16h ld_frag_a(const _Float16* A, int lda, int mbase,
                                        int kbase, int lane) {
  int m  = mbase + (lane & 15);
  int hi = (lane >> 4) & 1;
  const _Float16* p = A + (size_t)m * lda + kbase + hi * 8;
  V16 r;
  r.q[0] = *(const uint4*)(p);
  r.q[1] = *(const uint4*)(p + 16);
  return r.h;
}

// B fragment: 32x16 f16 (KxN) taken from an (N x K) row-major source
// (i.e. weight stored (out,in)).  lane l: col n = nbase + (l&15); hi=l>>4;
// element j<8 -> k = kbase + hi*8 + j, j>=8 -> k = kbase + hi*8 + 16 + (j-8).
static __device__ inline v16h ld_frag_b(const _Float16* Bm, int ldb, int nbase,
                                        int kbase, int lane) {
  int n  = nbase + (lane & 15);
  int hi = (lane >> 4) & 1;
  const _Float16* p = Bm + (size_t)n * ldb + kbase + hi * 8;
  V16 r;
  r.q[0] = *(const uint4*)(p);
  r.q[1] = *(const uint4*)(p + 16);
  return r.h;
}

static __device__ inline v8f wmma16(v16h a, v16h b, v8f c) {
  return __builtin_amdgcn_wmma_f32_16x16x32_f16(false, a, false, b,
                                                (short)0, c, false, false);
}

static __device__ inline float wave_sum(float v) {
  #pragma unroll
  for (int m = 16; m >= 1; m >>= 1) v += __shfl_xor(v, m, 32);
  return v;
}

// ---------------- residual base: out = x_in ----------------
__global__ void k_copy(const float* __restrict__ x, float* __restrict__ o, int n) {
  int g = blockIdx.x * blockDim.x + threadIdx.x;
  if (g < n) o[g] = x[g];
}

// ---------------- router ----------------
__global__ void k_mean(const float* __restrict__ x, float* __restrict__ xm) {
  int g = blockIdx.x * blockDim.x + threadIdx.x;           // (b,c,p)
  if (g >= Bsz * Cc * Nn_TOK) return;
  int p = g & (Nn_TOK - 1);
  int c = (g >> 10) & (Cc - 1);
  int b = g >> 17;
  float a = 0.f;
  #pragma unroll
  for (int t = 0; t < Tt; ++t)
    a += x[(size_t)((b * Tt + t) * Cc + c) * Nn_TOK + p];
  xm[g] = a * (1.f / Tt);
}

__global__ void k_conv3(const float* __restrict__ xm, const float* __restrict__ w,
                        const float* __restrict__ bb, float* __restrict__ h1) {
  int g = blockIdx.x * blockDim.x + threadIdx.x;           // (b,o,p) 8*32*1024
  if (g >= Bsz * 32 * Nn_TOK) return;
  int p = g & (Nn_TOK - 1);
  int o = (g >> 10) & 31;
  int b = g >> 15;
  int y = p >> 5, x = p & 31;
  float acc = bb[o];
  for (int c = 0; c < Cc; ++c) {
    const float* xp = xm + (size_t)(b * Cc + c) * Nn_TOK;
    const float* wp = w + (size_t)(o * Cc + c) * 9;
    #pragma unroll
    for (int ky = 0; ky < 3; ++ky) {
      int yy = y + ky - 1;
      if (yy < 0 || yy > 31) continue;
      #pragma unroll
      for (int kx = 0; kx < 3; ++kx) {
        int xx = x + kx - 1;
        if (xx < 0 || xx > 31) continue;
        acc += xp[yy * 32 + xx] * wp[ky * 3 + kx];
      }
    }
  }
  h1[g] = (acc >= 0.f) ? acc : 0.01f * acc;                // LeakyReLU
}

__global__ void k_score(const float* __restrict__ h1, const float* __restrict__ w2,
                        const float* __restrict__ b2, float* __restrict__ sc) {
  int g = blockIdx.x * blockDim.x + threadIdx.x;           // (b,p)
  if (g >= Bsz * Nn_TOK) return;
  int p = g & (Nn_TOK - 1);
  int b = g >> 10;
  float acc = b2[0];
  #pragma unroll
  for (int o = 0; o < 32; ++o)
    acc += h1[(size_t)(b * 32 + o) * Nn_TOK + p] * w2[o];
  sc[g] = 1.f / (1.f + __expf(-acc));
}

// one block (512 thr) per batch: bitonic sort 1024 (score,idx) descending,
// then sort kept 512 indices ascending.
__global__ void k_topk(const float* __restrict__ sc, int* __restrict__ idx) {
  __shared__ float s[1024];
  __shared__ int   id[1024];
  int b = blockIdx.x, t = threadIdx.x;
  s[t] = sc[b * Nn_TOK + t];               id[t] = t;
  s[t + 512] = sc[b * Nn_TOK + t + 512];   id[t + 512] = t + 512;
  __syncthreads();
  for (int kk = 2; kk <= 1024; kk <<= 1)
    for (int j = kk >> 1; j > 0; j >>= 1) {
      #pragma unroll
      for (int base = 0; base < 1024; base += 512) {
        int i = t + base, ixj = i ^ j;
        if (ixj > i) {
          bool up = ((i & kk) == 0);
          float si = s[i], sj = s[ixj];
          int ii = id[i], ij = id[ixj];
          bool better = (si > sj) || (si == sj && ii < ij);
          bool sw = up ? (!better) : better;
          if (sw) { s[i] = sj; s[ixj] = si; id[i] = ij; id[ixj] = ii; }
        }
      }
      __syncthreads();
    }
  // ascending sort of kept ids
  for (int kk = 2; kk <= 512; kk <<= 1)
    for (int j = kk >> 1; j > 0; j >>= 1) {
      int i = t, ixj = i ^ j;
      if (ixj > i && ixj < 512) {
        bool up = ((i & kk) == 0);
        int a = id[i], c = id[ixj];
        bool sw = up ? (a > c) : (a < c);
        if (sw) { id[i] = c; id[ixj] = a; }
      }
      __syncthreads();
    }
  idx[b * Kk_TOK + t] = id[t];
}

// ---------------- gather + STE + rmsnorm -> f16 ----------------
__global__ void k_gather_rms(const float* __restrict__ x, const int* __restrict__ idx,
                             const float* __restrict__ sc, const float* __restrict__ nw,
                             _Float16* __restrict__ z) {
  int wid  = blockIdx.x * (blockDim.x >> 5) + (threadIdx.x >> 5);
  int lane = threadIdx.x & 31;
  if (wid >= ROWS) return;
  int t = wid & 7, bk = wid >> 3, kk = bk & 511, b = bk >> 9;
  int p = idx[b * Kk_TOK + kk];
  float s = sc[b * Nn_TOK + p];
  float ste = s / (s + 1e-6f);
  const float* src = x + (size_t)((b * Tt + t) * Cc) * Nn_TOK + p;
  float v[4], ss = 0.f;
  #pragma unroll
  for (int j = 0; j < 4; ++j) {
    int c = lane + j * 32;
    v[j] = src[(size_t)c * Nn_TOK] * ste;
    ss += v[j] * v[j];
  }
  ss = wave_sum(ss);
  float inv = rsqrtf(ss * (1.f / Cc) + 1e-5f);
  #pragma unroll
  for (int j = 0; j < 4; ++j) {
    int c = lane + j * 32;
    z[(size_t)wid * Cc + c] = (_Float16)(v[j] * inv * nw[c]);
  }
}

// ---------------- weight conversions ----------------
__global__ void k_f2h(const float* __restrict__ s, _Float16* __restrict__ d, int n) {
  int g = blockIdx.x * blockDim.x + threadIdx.x;
  if (g < n) d[g] = (_Float16)s[g];
}
__global__ void k_padw(const float* __restrict__ s, _Float16* __restrict__ d) {
  int g = blockIdx.x * blockDim.x + threadIdx.x;
  if (g >= XPAD * DI) return;
  int row = g >> 8;                                        // /256
  d[g] = (row < (DTR + 2 * DS)) ? (_Float16)s[g] : (_Float16)0.f;
}

// ---------------- generic WMMA GEMM with TDM-staged B tile ----------------
// C(MxN) = A(MxK) * B(NxK)^T + bias.  One block = one n-tile (16 cols) x
// eight m-tiles (128 rows).  Wave 0 DMAs the 16xK f16 B tile into LDS with
// the Tensor Data Mover, waits on TENSORcnt, then all 8 waves consume it
// via ds_load fragments while streaming A from global.
__global__ void k_gemm_f16(const _Float16* __restrict__ A, const _Float16* __restrict__ Bw,
                           const float* __restrict__ bias, float* __restrict__ C,
                           int M, int Nld, int Kd, int tiles_n) {
  __shared__ _Float16 Btile[16 * 256];                     // Kd <= 256 -> 8 KB
  int wid = threadIdx.x >> 5, lane = threadIdx.x & 31;
  int tn  = blockIdx.x % tiles_n;
  int tmb = blockIdx.x / tiles_n;
  if (wid == 0) {
    // Build 2D tensor descriptor (ISA 08_async_tensor 8.3/8.4):
    unsigned long long ga = (unsigned long long)(const void*)(Bw + (size_t)tn * 16 * Kd);
    unsigned lds = (unsigned)(unsigned long long)(const void*)&Btile[0];
    unsigned dim0 = (unsigned)Kd;                 // row length (elements)
    unsigned dim1 = (unsigned)(tiles_n * 16);     // number of rows
    unsigned long long str0 = (unsigned long long)Kd;
    v4u g0; v8i g1; v4i g2, g3; v8i g4;
    g0[0] = 1u;                                   // count=1, user descriptor
    g0[1] = lds;                                  // lds_addr (bytes)
    g0[2] = (unsigned)(ga & 0xffffffffu);         // global_addr[31:0]
    g0[3] = (unsigned)((ga >> 32) & 0x01ffffffu) | (2u << 30);  // [56:32] | type=2
    g1[0] = (int)(1u << 16);                      // data_size=1 (2 bytes)
    g1[1] = (int)((dim0 & 0xffffu) << 16);        // tensor_dim0[15:0]
    g1[2] = (int)((dim0 >> 16) | ((dim1 & 0xffffu) << 16));   // dim0 hi | dim1 lo
    g1[3] = (int)((dim1 >> 16) | ((unsigned)Kd << 16));       // dim1 hi | tile_dim0
    g1[4] = (int)16;                              // tile_dim1=16, tile_dim2=0
    g1[5] = (int)(unsigned)(str0 & 0xffffffffu);  // tensor_dim0_stride lo
    g1[6] = (int)(unsigned)((str0 >> 32) & 0xffffu);          // stride hi | stride1 lo
    g1[7] = 0;
    g2[0] = 0; g2[1] = 0; g2[2] = 0; g2[3] = 0;
    g3[0] = 0; g3[1] = 0; g3[2] = 0; g3[3] = 0;
    g4[0] = 0; g4[1] = 0; g4[2] = 0; g4[3] = 0;
    g4[4] = 0; g4[5] = 0; g4[6] = 0; g4[7] = 0;
    __builtin_amdgcn_tensor_load_to_lds(g0, g1, g2, g3, g4, 0);
    __builtin_amdgcn_s_wait_tensorcnt(0);
  }
  __syncthreads();
  int tm = tmb * 8 + wid;
  if (tm * 16 >= M) return;                                // wave-uniform
  v8f acc = {};
  for (int k0 = 0; k0 < Kd; k0 += 32) {
    __builtin_prefetch(A + (size_t)(tm * 16 + (lane & 15)) * Kd + k0 + 64, 0, 0);
    v16h a = ld_frag_a(A, Kd, tm * 16, k0, lane);
    v16h b = ld_frag_b(Btile, Kd, 0, k0, lane);            // from LDS
    acc = wmma16(a, b, acc);
  }
  int n = tn * 16 + (lane & 15);
  int mb = tm * 16 + ((lane >> 4) & 1) * 8;
  float bv = bias ? bias[n] : 0.f;
  #pragma unroll
  for (int r = 0; r < 8; ++r)
    C[(size_t)(mb + r) * Nld + n] = acc[r] + bv;
}

// ---------------- depthwise causal conv4 + silu ----------------
__global__ void k_conv_silu(const float* __restrict__ xz, const float* __restrict__ w,
                            const float* __restrict__ b, float* __restrict__ xc,
                            _Float16* __restrict__ xch) {
  int g = blockIdx.x * blockDim.x + threadIdx.x;           // (bk,d) 4096*256
  if (g >= (ROWS / Tt) * DI) return;
  int d = g & (DI - 1), bk = g >> 8;
  float w0 = w[d * 4 + 0], w1 = w[d * 4 + 1], w2 = w[d * 4 + 2], w3 = w[d * 4 + 3];
  float bd = b[d];
  float x0 = 0.f, x1 = 0.f, x2 = 0.f;
  #pragma unroll
  for (int t = 0; t < Tt; ++t) {
    size_t row = (size_t)bk * Tt + t;
    float x3 = xz[row * (2 * DI) + d];
    float a = w0 * x0 + w1 * x1 + w2 * x2 + w3 * x3 + bd;
    float v = a / (1.f + __expf(-a));                      // silu
    xc[row * DI + d]  = v;
    xch[row * DI + d] = (_Float16)v;
    x0 = x1; x1 = x2; x2 = x3;
  }
}

// ---------------- dt = softplus(x_dbl[:,0:8] @ dtw^T + dtb) ----------------
__global__ void k_dt(const float* __restrict__ xd, const float* __restrict__ dtw,
                     const float* __restrict__ dtb, float* __restrict__ dtf) {
  int g = blockIdx.x * blockDim.x + threadIdx.x;           // (row,d) 32768*256
  if (g >= ROWS * DI) return;
  int d = g & (DI - 1), row = g >> 8;
  float acc = dtb[d];
  #pragma unroll
  for (int r = 0; r < DTR; ++r)
    acc += xd[(size_t)row * XPAD + r] * dtw[d * DTR + r];
  dtf[g] = (acc > 20.f) ? acc : __logf(1.f + __expf(acc));
}

// ---------------- selective scan + gate -> y f16 ----------------
__global__ void k_scan(const float* __restrict__ xz, const float* __restrict__ xc,
                       const float* __restrict__ dtf, const float* __restrict__ xd,
                       const float* __restrict__ A_log, const float* __restrict__ Dp,
                       _Float16* __restrict__ y16) {
  int g = blockIdx.x * blockDim.x + threadIdx.x;           // (bk,d)
  if (g >= (ROWS / Tt) * DI) return;
  int d = g & (DI - 1), bk = g >> 8;
  float Af[DS], h[DS];
  #pragma unroll
  for (int s = 0; s < DS; ++s) { Af[s] = -__expf(A_log[d * DS + s]); h[s] = 0.f; }
  float Dv = Dp[d];
  #pragma unroll
  for (int t = 0; t < Tt; ++t) {
    size_t row = (size_t)bk * Tt + t;
    float dt = dtf[row * DI + d];
    float u  = xc[row * DI + d];
    float du = dt * u;
    const float* bc = xd + row * XPAD;
    float y = 0.f;
    #pragma unroll
    for (int s = 0; s < DS; ++s) {
      h[s] = h[s] * __expf(dt * Af[s]) + du * bc[DTR + s];
      y += h[s] * bc[DTR + DS + s];
    }
    y += u * Dv;
    float zg = xz[row * (2 * DI) + DI + d];
    y *= zg / (1.f + __expf(-zg));                         // * silu(z)
    y16[row * DI + d] = (_Float16)y;
  }
}

// ---------------- reorder mamba output to attention order (f16) ----------
__global__ void k_z2att(const float* __restrict__ z2, _Float16* __restrict__ za) {
  int g = blockIdx.x * blockDim.x + threadIdx.x;
  if (g >= ROWS * Cc) return;
  int c = g & (Cc - 1);
  int ra = g >> 7;
  int kk = ra & 511, bt = ra >> 9, t = bt & 7, b = bt >> 3;
  size_t rm = (size_t)(b * Kk_TOK + kk) * Tt + t;
  za[g] = (_Float16)z2[rm * Cc + c];
}

// ---------------- split qkv -> per-(bt,h) f16 Q,K and transposed V --------
__global__ void k_qkvsplit(const float* __restrict__ qkv, _Float16* __restrict__ q,
                           _Float16* __restrict__ k, _Float16* __restrict__ vt) {
  int g = blockIdx.x * blockDim.x + threadIdx.x;
  if (g >= ROWS * Cc) return;
  int c = g & (Cc - 1);
  int ra = g >> 7;
  int tok = ra & 511, bt = ra >> 9;
  int h = c >> 5, dh = c & 31;
  size_t base = (size_t)ra * (3 * Cc);
  size_t qi = ((size_t)(bt * NH + h) * Kk_TOK + tok) * DH + dh;
  q[qi] = (_Float16)qkv[base + c];
  k[qi] = (_Float16)qkv[base + Cc + c];
  vt[((size_t)(bt * NH + h) * DH + dh) * Kk_TOK + tok] = (_Float16)qkv[base + 2 * Cc + c];
}

// ---------------- flash attention, one wave per (bt,h,16-row q tile) ------
__global__ void k_attn(const _Float16* __restrict__ Q, const _Float16* __restrict__ Km,
                       const _Float16* __restrict__ Vt, _Float16* __restrict__ O) {
  __shared__ _Float16 Plds[8][16 * 32];
  int wid = threadIdx.x >> 5, lane = threadIdx.x & 31;
  int task = blockIdx.x * 8 + wid;
  if (task >= BT * NH * (Kk_TOK / 16)) return;             // wave-uniform
  int qt = task & 31, h = (task >> 5) & 3, bt = task >> 7;
  _Float16* pl = Plds[wid];
  const _Float16* q  = Q  + (size_t)(bt * NH + h) * Kk_TOK * DH;
  const _Float16* km = Km + (size_t)(bt * NH + h) * Kk_TOK * DH;
  const _Float16* vt = Vt + (size_t)(bt * NH + h) * DH * Kk_TOK;
  v16h qa = ld_frag_a(q, DH, qt * 16, 0, lane);
  v8f o0 = {}, o1 = {};
  float mi[8], li[8];
  #pragma unroll
  for (int r = 0; r < 8; ++r) { mi[r] = -1e30f; li[r] = 0.f; }
  const float sc = 0.17677669529663687f;                   // 1/sqrt(32)
  int mrow0 = ((lane >> 4) & 1) * 8;
  int ncol  = lane & 15;
  for (int kb = 0; kb < Kk_TOK; kb += 32) {
    v16h kb0 = ld_frag_b(km, DH, kb, 0, lane);
    v16h kb1 = ld_frag_b(km, DH, kb + 16, 0, lane);
    v8f s0 = {}, s1 = {};
    s0 = wmma16(qa, kb0, s0);
    s1 = wmma16(qa, kb1, s1);
    #pragma unroll
    for (int r = 0; r < 8; ++r) {
      float a0 = s0[r] * sc, a1 = s1[r] * sc;
      float mx = fmaxf(a0, a1);
      #pragma unroll
      for (int d = 8; d >= 1; d >>= 1) mx = fmaxf(mx, __shfl_xor(mx, d, 32));
      float mnew = fmaxf(mi[r], mx);
      float corr = __expf(mi[r] - mnew);
      float p0 = __expf(a0 - mnew), p1 = __expf(a1 - mnew);
      float rs = p0 + p1;
      #pragma unroll
      for (int d = 8; d >= 1; d >>= 1) rs += __shfl_xor(rs, d, 32);
      li[r] = li[r] * corr + rs;
      mi[r] = mnew;
      o0[r] = o0[r] * corr;
      o1[r] = o1[r] * corr;
      int mr = mrow0 + r;
      pl[mr * 32 + ncol]      = (_Float16)p0;
      pl[mr * 32 + 16 + ncol] = (_Float16)p1;
    }
    v16h pa  = ld_frag_a(pl, 32, 0, 0, lane);              // LDS round-trip reshape
    v16h vb0 = ld_frag_b(vt, Kk_TOK, 0, kb, lane);
    v16h vb1 = ld_frag_b(vt, Kk_TOK, 16, kb, lane);
    o0 = wmma16(pa, vb0, o0);
    o1 = wmma16(pa, vb1, o1);
  }
  int tokb = qt * 16 + mrow0;
  #pragma unroll
  for (int r = 0; r < 8; ++r) {
    float inv = 1.f / li[r];
    size_t row = (size_t)(bt * Kk_TOK + tokb + r) * Cc + h * DH;
    O[row + ncol]      = (_Float16)(o0[r] * inv);
    O[row + 16 + ncol] = (_Float16)(o1[r] * inv);
  }
}

// ---------------- layernorm(zs + o) and scatter-add into d_out ------------
__global__ void k_ln_scatter(const float* __restrict__ z2, const float* __restrict__ ao,
                             const int* __restrict__ idx, const float* __restrict__ lw,
                             const float* __restrict__ lb, float* __restrict__ out) {
  int wid  = blockIdx.x * (blockDim.x >> 5) + (threadIdx.x >> 5);
  int lane = threadIdx.x & 31;
  if (wid >= ROWS) return;
  int tok = wid & 511, bt = wid >> 9, t = bt & 7, b = bt >> 3;
  int p = idx[b * Kk_TOK + tok];
  size_t rm = (size_t)(b * Kk_TOK + tok) * Tt + t;
  float v[4], mu = 0.f;
  #pragma unroll
  for (int j = 0; j < 4; ++j) {
    int c = lane + j * 32;
    v[j] = z2[rm * Cc + c] + ao[(size_t)wid * Cc + c];
    mu += v[j];
  }
  mu = wave_sum(mu) * (1.f / Cc);
  float var = 0.f;
  #pragma unroll
  for (int j = 0; j < 4; ++j) { float d = v[j] - mu; var += d * d; }
  var = wave_sum(var) * (1.f / Cc);
  float inv = rsqrtf(var + 1e-5f);
  #pragma unroll
  for (int j = 0; j < 4; ++j) {
    int c = lane + j * 32;
    float zo = (v[j] - mu) * inv * lw[c] + lb[c];
    out[(size_t)((b * Tt + t) * Cc + c) * Nn_TOK + p] += zo;
  }
}

// ================================================================
extern "C" void kernel_launch(void* const* d_in, const int* in_sizes, int n_in,
                              void* d_out, int out_size, void* d_ws, size_t ws_size,
                              hipStream_t stream) {
  (void)in_sizes; (void)n_in; (void)out_size; (void)ws_size;
  const float* x_in     = (const float*)d_in[0];
  const float* norm1_w  = (const float*)d_in[1];
  const float* in_proj  = (const float*)d_in[2];
  const float* conv_w   = (const float*)d_in[3];
  const float* conv_b   = (const float*)d_in[4];
  const float* xproj_w  = (const float*)d_in[5];
  const float* dtproj_w = (const float*)d_in[6];
  const float* dtproj_b = (const float*)d_in[7];
  const float* A_log    = (const float*)d_in[8];
  const float* Dp       = (const float*)d_in[9];
  const float* outproj  = (const float*)d_in[10];
  const float* r1_w     = (const float*)d_in[11];
  const float* r1_b     = (const float*)d_in[12];
  const float* r2_w     = (const float*)d_in[13];
  const float* r2_b     = (const float*)d_in[14];
  const float* attin_w  = (const float*)d_in[15];
  const float* attin_b  = (const float*)d_in[16];
  const float* attout_w = (const float*)d_in[17];
  const float* attout_b = (const float*)d_in[18];
  const float* ln_w     = (const float*)d_in[19];
  const float* ln_b     = (const float*)d_in[20];
  float* out = (float*)d_out;

  char* w = (char*)d_ws;
  size_t off = 0;
  auto take = [&](size_t bytes) { size_t r = off; off += (bytes + 255) & ~(size_t)255; return r; };
  float*     xm     = (float*)(w + take((size_t)Bsz * Cc * Nn_TOK * 4));
  float*     h1     = (float*)(w + take((size_t)Bsz * 32 * Nn_TOK * 4));
  float*     scores = (float*)(w + take((size_t)Bsz * Nn_TOK * 4));
  int*       idx    = (int*)  (w + take((size_t)Bsz * Kk_TOK * 4));
  _Float16*  zf16   = (_Float16*)(w + take((size_t)ROWS * Cc * 2));
  _Float16*  w_inp  = (_Float16*)(w + take((size_t)2 * DI * Cc * 2));
  _Float16*  w_xp   = (_Float16*)(w + take((size_t)XPAD * DI * 2));
  _Float16*  w_outp = (_Float16*)(w + take((size_t)Cc * DI * 2));
  _Float16*  w_ain  = (_Float16*)(w + take((size_t)3 * Cc * Cc * 2));
  _Float16*  w_aout = (_Float16*)(w + take((size_t)Cc * Cc * 2));
  float*     xz     = (float*)(w + take((size_t)ROWS * 2 * DI * 4));
  float*     xc     = (float*)(w + take((size_t)ROWS * DI * 4));
  _Float16*  xcf16  = (_Float16*)(w + take((size_t)ROWS * DI * 2));
  float*     xdbl   = (float*)(w + take((size_t)ROWS * XPAD * 4));
  float*     dtf    = (float*)(w + take((size_t)ROWS * DI * 4));
  _Float16*  yf16   = (_Float16*)(w + take((size_t)ROWS * DI * 2));
  float*     z2     = (float*)(w + take((size_t)ROWS * Cc * 4));
  _Float16*  z2a    = (_Float16*)(w + take((size_t)ROWS * Cc * 2));
  float*     qkv    = (float*)(w + take((size_t)ROWS * 3 * Cc * 4));
  _Float16*  qf     = (_Float16*)(w + take((size_t)ROWS * Cc * 2));
  _Float16*  kf     = (_Float16*)(w + take((size_t)ROWS * Cc * 2));
  _Float16*  vtf    = (_Float16*)(w + take((size_t)ROWS * Cc * 2));
  _Float16*  of16   = (_Float16*)(w + take((size_t)ROWS * Cc * 2));
  float*     ao     = (float*)(w + take((size_t)ROWS * Cc * 4));

  const int TB = 256;
  auto grid1 = [](long n, int tb) { return (int)((n + tb - 1) / tb); };

  // residual base
  k_copy<<<grid1((long)Bsz * Tt * Cc * Nn_TOK, TB), TB, 0, stream>>>(
      x_in, out, Bsz * Tt * Cc * Nn_TOK);
  // router
  k_mean<<<grid1((long)Bsz * Cc * Nn_TOK, TB), TB, 0, stream>>>(x_in, xm);
  k_conv3<<<grid1((long)Bsz * 32 * Nn_TOK, TB), TB, 0, stream>>>(xm, r1_w, r1_b, h1);
  k_score<<<grid1((long)Bsz * Nn_TOK, TB), TB, 0, stream>>>(h1, r2_w, r2_b, scores);
  k_topk<<<Bsz, 512, 0, stream>>>(scores, idx);
  // gather + rmsnorm
  k_gather_rms<<<ROWS / 8, TB, 0, stream>>>(x_in, idx, scores, norm1_w, zf16);
  // weight conversion
  k_f2h<<<grid1(2 * DI * Cc, TB), TB, 0, stream>>>(in_proj, w_inp, 2 * DI * Cc);
  k_padw<<<grid1(XPAD * DI, TB), TB, 0, stream>>>(xproj_w, w_xp);
  k_f2h<<<grid1(Cc * DI, TB), TB, 0, stream>>>(outproj, w_outp, Cc * DI);
  k_f2h<<<grid1(3 * Cc * Cc, TB), TB, 0, stream>>>(attin_w, w_ain, 3 * Cc * Cc);
  k_f2h<<<grid1(Cc * Cc, TB), TB, 0, stream>>>(attout_w, w_aout, Cc * Cc);

  const int tiles_m = ROWS / 16;                    // 2048, 8 per block
  // mamba: in_proj GEMM (32768x512x128)
  {
    int tn = (2 * DI) / 16;
    k_gemm_f16<<<tn * (tiles_m / 8), TB, 0, stream>>>(
        zf16, w_inp, nullptr, xz, ROWS, 2 * DI, Cc, tn);
  }
  k_conv_silu<<<grid1((long)(ROWS / Tt) * DI, TB), TB, 0, stream>>>(
      xz, conv_w, conv_b, xc, xcf16);
  // x_proj GEMM (32768x48x256)
  {
    int tn = XPAD / 16;
    k_gemm_f16<<<tn * (tiles_m / 8), TB, 0, stream>>>(
        xcf16, w_xp, nullptr, xdbl, ROWS, XPAD, DI, tn);
  }
  k_dt<<<grid1((long)ROWS * DI, TB), TB, 0, stream>>>(xdbl, dtproj_w, dtproj_b, dtf);
  k_scan<<<grid1((long)(ROWS / Tt) * DI, TB), TB, 0, stream>>>(
      xz, xc, dtf, xdbl, A_log, Dp, yf16);
  // out_proj GEMM (32768x128x256)
  {
    int tn = Cc / 16;
    k_gemm_f16<<<tn * (tiles_m / 8), TB, 0, stream>>>(
        yf16, w_outp, nullptr, z2, ROWS, Cc, DI, tn);
  }
  // attention
  k_z2att<<<grid1((long)ROWS * Cc, TB), TB, 0, stream>>>(z2, z2a);
  {
    int tn = (3 * Cc) / 16;
    k_gemm_f16<<<tn * (tiles_m / 8), TB, 0, stream>>>(
        z2a, w_ain, attin_b, qkv, ROWS, 3 * Cc, Cc, tn);
  }
  k_qkvsplit<<<grid1((long)ROWS * Cc, TB), TB, 0, stream>>>(qkv, qf, kf, vtf);
  k_attn<<<(BT * NH * (Kk_TOK / 16)) / 8, TB, 0, stream>>>(qf, kf, vtf, of16);
  {
    int tn = Cc / 16;
    k_gemm_f16<<<tn * (tiles_m / 8), TB, 0, stream>>>(
        of16, w_aout, attout_b, ao, ROWS, Cc, Cc, tn);
  }
  // layernorm + scatter
  k_ln_scatter<<<ROWS / 8, TB, 0, stream>>>(z2, ao, idx, ln_w, ln_b, out);
}
